// SimpleAttn_84327387890091
// MI455X (gfx1250) — matrix operand order repaired and checked by reading, not compiled, not run on hardware
//
#include <hip/hip_runtime.h>

// ---------------------------------------------------------------------------
// CDNA5 (gfx1250) WMMA implementation of SimpleAttn.
//   - All matmuls through v_wmma_f32_16x16x32_f16 (wave32, f32 accum)
//   - Double-buffered LDS staging via GLOBAL_LOAD_ASYNC_TO_LDS_B128 (ASYNCcnt)
//     when the builtin is available; sync copy fallback.
//   - Block tile 128x128, 8 waves, each wave 32x64 (8 WMMAs / K-step)
// ---------------------------------------------------------------------------

typedef __attribute__((ext_vector_type(16))) _Float16 v16h;
typedef __attribute__((ext_vector_type(8)))  float    v8f;
typedef __attribute__((ext_vector_type(4)))  int      v4i;

#define LDS_STRIDE 40   // padded row stride (halfs); 80B = 5*16B -> 16B aligned
#define TILE_HALFS (128 * LDS_STRIDE)

#if __has_builtin(__builtin_amdgcn_global_load_async_to_lds_b128)
#define USE_ASYNC_LDS 1
#else
#define USE_ASYNC_LDS 0
#endif

typedef __attribute__((address_space(1))) v4i* g_b128_ptr;
typedef __attribute__((address_space(3))) v4i* l_b128_ptr;

union Frag16 { v16h v; uint4 q[2]; };
union H8     { uint4 q; _Float16 h[8]; };

// 16-byte staging copy: async DMA into LDS when available.
__device__ __forceinline__ void copy16(_Float16* lds_dst, const _Float16* g_src) {
#if USE_ASYNC_LDS
  __builtin_amdgcn_global_load_async_to_lds_b128(
      (g_b128_ptr)g_src, (l_b128_ptr)lds_dst, /*offset=*/0, /*cpol=*/0);
#else
  *(uint4*)lds_dst = *(const uint4*)g_src;
#endif
}

__device__ __forceinline__ void wait_stage() {
#if USE_ASYNC_LDS
#if __has_builtin(__builtin_amdgcn_s_wait_asynccnt)
  __builtin_amdgcn_s_wait_asynccnt(0);
#else
  asm volatile("s_wait_asynccnt 0x0" ::: "memory");
#endif
#endif
}

// A fragment (16x32 f16, M x K): lane = M (mod 16); lanes 0-15 hold K 0-7 and
// 16-23, lanes 16-31 hold K 8-15 and 24-31 (2 halfs per VGPR).
__device__ __forceinline__ v16h load_a_frag(const _Float16* lds, int row, int halfsel) {
  Frag16 f;
  const _Float16* p = lds + row * LDS_STRIDE + halfsel * 8;
  f.q[0] = *(const uint4*)(p);        // K = halfsel*8 .. +7
  f.q[1] = *(const uint4*)(p + 16);   // K = 16 + halfsel*8 .. +7
  return f.v;
}

// B fragment (32x16 f16, K x N), LDS tile stored as [N][K] (B^T rows):
// lane = N (mod 16); lanes 0-15 hold K 0-7 and 16-23, lanes 16-31 hold K 8-15 and 24-31.
__device__ __forceinline__ v16h load_b_frag(const _Float16* lds, int nrow, int halfsel) {
  Frag16 f;
  const _Float16* p = lds + nrow * LDS_STRIDE + halfsel * 8;
  f.q[0] = *(const uint4*)(p);
  f.q[1] = *(const uint4*)(p + 16);
  return f.v;
}

// C = act(A @ B^T + bias)
//   A: (M,K) f16 row-major
//   B: if !B_IS_KN: (N,K) f16 row-major; if B_IS_KN: (K,N) f16 row-major
//      (transposed during LDS fill; sync path only)
//   Block tile 128(M) x 128(N), 8 waves; wave tile 32(M) x 64(N).
template <bool RELU, bool HAS_BIAS, bool B_IS_KN, bool OUT_HALF>
__global__ __launch_bounds__(256)
void gemm_wmma(const _Float16* __restrict__ A, int lda,
               const _Float16* __restrict__ B, int ldb,
               const float* __restrict__ bias,
               void* __restrict__ Cout, int ldc,
               int M, int N, int K)
{
  __shared__ _Float16 sA[2 * TILE_HALFS];
  __shared__ _Float16 sB[2 * TILE_HALFS];

  const int t       = threadIdx.x;
  const int lane    = t & 31;
  const int wave    = t >> 5;
  const int halfsel = lane >> 4;
  const int lr      = lane & 15;
  const int wm      = wave >> 1;   // 0..3 : 32-row strip
  const int wn      = wave & 1;    // 0..1 : 64-col strip
  const int m0      = blockIdx.y * 128;
  const int n0      = blockIdx.x * 128;

  // Stage one 128x32 A tile + 128x32 B tile (as [N][K]) into buffer `buf`.
  auto stage = [&](int k0, int buf) {
    _Float16* dA = sA + buf * TILE_HALFS;
    _Float16* dB = sB + buf * TILE_HALFS;
    #pragma unroll
    for (int j = 0; j < 2; ++j) {             // A: 512 x 16B chunks
      int u = t + 256 * j;
      int row = u >> 2, kc = u & 3;
      copy16(dA + row * LDS_STRIDE + kc * 8,
             A + (size_t)(m0 + row) * lda + k0 + kc * 8);
    }
    if (!B_IS_KN) {
      #pragma unroll
      for (int j = 0; j < 2; ++j) {           // B^T rows: 512 x 16B chunks
        int u = t + 256 * j;
        int n = u >> 2, kc = u & 3;
        copy16(dB + n * LDS_STRIDE + kc * 8,
               B + (size_t)(n0 + n) * ldb + k0 + kc * 8);
      }
    } else {
      // B is (K,N): load 8 halfs along N, scatter transposed into LDS.
      #pragma unroll
      for (int j = 0; j < 2; ++j) {
        int u  = t + 256 * j;                 // 0..511
        int k  = u & 31;                      // 0..31
        int nc = (u >> 5) * 8;                // 0..120
        H8 d; d.q = *(const uint4*)(B + (size_t)(k0 + k) * ldb + n0 + nc);
        #pragma unroll
        for (int i = 0; i < 8; ++i) dB[(nc + i) * LDS_STRIDE + k] = d.h[i];
      }
    }
  };

  v8f acc[2][4];
  #pragma unroll
  for (int i = 0; i < 2; ++i)
    #pragma unroll
    for (int j = 0; j < 4; ++j)
      #pragma unroll
      for (int e = 0; e < 8; ++e) acc[i][j][e] = 0.0f;

  const int nsteps = K >> 5;     // K / 32
  stage(0, 0);

  for (int s = 0; s < nsteps; ++s) {
    const int buf = s & 1;
    wait_stage();                // own async copies into buf complete
    __syncthreads();             // everyone's copies visible

    if (s + 1 < nsteps) stage((s + 1) * 32, buf ^ 1);   // overlap next tile

    const _Float16* cA = sA + buf * TILE_HALFS;
    const _Float16* cB = sB + buf * TILE_HALFS;

    v16h af[2], bf[4];
    #pragma unroll
    for (int mi = 0; mi < 2; ++mi)
      af[mi] = load_a_frag(cA, wm * 32 + mi * 16 + lr, halfsel);
    #pragma unroll
    for (int ni = 0; ni < 4; ++ni)
      bf[ni] = load_b_frag(cB, wn * 64 + ni * 16 + lr, halfsel);

    #pragma unroll
    for (int mi = 0; mi < 2; ++mi)
      #pragma unroll
      for (int ni = 0; ni < 4; ++ni)
        acc[mi][ni] = __builtin_amdgcn_wmma_f32_16x16x32_f16(
            /*neg_a=*/false, af[mi], /*neg_b=*/false, bf[ni],
            /*c_mod=*/(short)0, acc[mi][ni],
            /*reuse_a=*/false, /*reuse_b=*/false);

    __syncthreads();             // all reads of buf done before it is reused
  }

  // ---- epilogue: bias / relu / store ----
  #pragma unroll
  for (int mi = 0; mi < 2; ++mi) {
    int rbase = m0 + wm * 32 + mi * 16 + halfsel * 8;
    #pragma unroll
    for (int ni = 0; ni < 4; ++ni) {
      int col = n0 + wn * 64 + ni * 16 + lr;
      float bv = HAS_BIAS ? bias[col] : 0.0f;
      #pragma unroll
      for (int r = 0; r < 8; ++r) {
        float v = acc[mi][ni][r] + bv;
        if (RELU) v = v > 0.0f ? v : 0.0f;
        size_t idx = (size_t)(rbase + r) * ldc + col;
        if (OUT_HALF) ((_Float16*)Cout)[idx] = (_Float16)v;
        else          ((float*)Cout)[idx]    = v;
      }
    }
  }
}

// ---------------------------------------------------------------------------
// f32 -> f16 conversion (4-wide)
// ---------------------------------------------------------------------------
typedef __attribute__((ext_vector_type(4))) _Float16 v4h;

__global__ void cvt_f32_to_f16(const float4* __restrict__ in,
                               v4h* __restrict__ out, int n4)
{
  int i = blockIdx.x * 256 + threadIdx.x;
  if (i < n4) {
    float4 f = in[i];
    v4h h;
    h[0] = (_Float16)f.x; h[1] = (_Float16)f.y;
    h[2] = (_Float16)f.z; h[3] = (_Float16)f.w;
    out[i] = h;
  }
}

// ---------------------------------------------------------------------------
// Row softmax: S (f32, ncols=2048) -> P (f16). One block of 256 per row.
// ---------------------------------------------------------------------------
__global__ __launch_bounds__(256)
void softmax_row(const float* __restrict__ S, _Float16* __restrict__ P, int ncols)
{
  __shared__ float red[256];
  const int row = blockIdx.x;
  const float* s = S + (size_t)row * ncols;
  _Float16*    p = P + (size_t)row * ncols;

  float vals[8];
  float m = -1e30f;
  #pragma unroll
  for (int j = 0; j < 8; ++j) {
    vals[j] = s[threadIdx.x + j * 256];
    m = fmaxf(m, vals[j]);
  }
  red[threadIdx.x] = m; __syncthreads();
  for (int off = 128; off > 0; off >>= 1) {
    if (threadIdx.x < off) red[threadIdx.x] = fmaxf(red[threadIdx.x], red[threadIdx.x + off]);
    __syncthreads();
  }
  m = red[0]; __syncthreads();

  float sum = 0.0f;
  #pragma unroll
  for (int j = 0; j < 8; ++j) { vals[j] = __expf(vals[j] - m); sum += vals[j]; }
  red[threadIdx.x] = sum; __syncthreads();
  for (int off = 128; off > 0; off >>= 1) {
    if (threadIdx.x < off) red[threadIdx.x] += red[threadIdx.x + off];
    __syncthreads();
  }
  float inv = 1.0f / red[0];
  #pragma unroll
  for (int j = 0; j < 8; ++j)
    p[threadIdx.x + j * 256] = (_Float16)(vals[j] * inv);
}

// ---------------------------------------------------------------------------
// Host orchestration
// ---------------------------------------------------------------------------
extern "C" void kernel_launch(void* const* d_in, const int* in_sizes, int n_in,
                              void* d_out, int out_size, void* d_ws, size_t ws_size,
                              hipStream_t stream)
{
  (void)in_sizes; (void)n_in; (void)out_size; (void)ws_size;

  const float* x   = (const float*)d_in[0];   // (8,2048,512)
  const float* We1 = (const float*)d_in[1];   // (2048,512)
  const float* be1 = (const float*)d_in[2];   // (2048)
  const float* We2 = (const float*)d_in[3];   // (512,2048)
  const float* be2 = (const float*)d_in[4];   // (512)
  const float* Wk1 = (const float*)d_in[5];   // (2048,512)
  const float* bk1 = (const float*)d_in[6];   // (2048)
  const float* Wk2 = (const float*)d_in[7];   // (256,2048)
  const float* bk2 = (const float*)d_in[8];   // (256)
  float* out = (float*)d_out;                 // (8,2048,512)

  const int B = 8, N = 2048, D = 512, H = 2048, DK = 128;
  const int BN = B * N;                        // 16384

  char* ws = (char*)d_ws;
  size_t off = 0;
  auto alloc = [&](size_t bytes) -> void* {
    void* p = ws + off;
    off += (bytes + 255) & ~(size_t)255;
    return p;
  };

  _Float16* x16    = (_Float16*)alloc((size_t)BN * D * 2);        // 16.8 MB
  _Float16* w1h    = (_Float16*)alloc((size_t)H * D * 2);         //  2.1 MB
  _Float16* w2h    = (_Float16*)alloc((size_t)D * H * 2);         //  2.1 MB
  _Float16* wk1h   = (_Float16*)alloc((size_t)H * D * 2);         //  2.1 MB
  _Float16* wk2h   = (_Float16*)alloc((size_t)(2 * DK) * H * 2);  //  1.0 MB
  _Float16* hbuf   = (_Float16*)alloc((size_t)BN * H * 2);        // 67.1 MB (reused)
  _Float16* embs16 = (_Float16*)alloc((size_t)BN * D * 2);        // 16.8 MB
  _Float16* kq16   = (_Float16*)alloc((size_t)BN * (2 * DK) * 2); //  8.4 MB
  float*    Sbuf   = (float*)   alloc((size_t)N * N * 4);         // 16.8 MB (per-batch reuse)
  _Float16* Pbuf   = (_Float16*)alloc((size_t)N * N * 2);         //  8.4 MB (per-batch reuse)

  // ---- f32 -> f16 conversions ----
  auto cvt = [&](const float* src, _Float16* dst, size_t n) {
    int n4 = (int)(n / 4);
    cvt_f32_to_f16<<<(n4 + 255) / 256, 256, 0, stream>>>(
        (const float4*)src, (v4h*)dst, n4);
  };
  cvt(x,   x16,  (size_t)BN * D);
  cvt(We1, w1h,  (size_t)H * D);
  cvt(We2, w2h,  (size_t)D * H);
  cvt(Wk1, wk1h, (size_t)H * D);
  cvt(Wk2, wk2h, (size_t)(2 * DK) * H);

  // ---- MLP path 1: embs ----
  {
    dim3 g(H / 128, BN / 128);  // h = relu(x @ We1^T + be1)  : (16384 x 2048)
    gemm_wmma<true, true, false, true><<<g, 256, 0, stream>>>(
        x16, D, w1h, D, be1, hbuf, H, BN, H, D);
  }
  {
    dim3 g(D / 128, BN / 128);  // embs = h @ We2^T + be2     : (16384 x 512)
    gemm_wmma<false, true, false, true><<<g, 256, 0, stream>>>(
        hbuf, H, w2h, H, be2, embs16, D, BN, D, H);
  }

  // ---- MLP path 2: kq ----
  {
    dim3 g(H / 128, BN / 128);  // h2 = relu(x @ Wk1^T + bk1)
    gemm_wmma<true, true, false, true><<<g, 256, 0, stream>>>(
        x16, D, wk1h, D, bk1, hbuf, H, BN, H, D);
  }
  {
    dim3 g((2 * DK) / 128, BN / 128);  // kq = h2 @ Wk2^T + bk2 : (16384 x 256)
    gemm_wmma<false, true, false, true><<<g, 256, 0, stream>>>(
        hbuf, H, wk2h, H, bk2, kq16, 2 * DK, BN, 2 * DK, H);
  }

  // ---- attention, one batch at a time (S/P scratch reused; stream-ordered) ----
  for (int b = 0; b < B; ++b) {
    const _Float16* Kb = kq16 + (size_t)b * N * (2 * DK);        // cols [0,128)
    const _Float16* Qb = Kb + DK;                                // cols [128,256)
    const _Float16* Vb = embs16 + (size_t)b * N * D;
    float*          Ob = out + (size_t)b * N * D;

    {
      dim3 g(N / 128, N / 128); // S = Q @ K^T : (2048 x 2048), K-dim 128
      gemm_wmma<false, false, false, false><<<g, 256, 0, stream>>>(
          Qb, 2 * DK, Kb, 2 * DK, nullptr, Sbuf, N, N, N, DK);
    }
    softmax_row<<<N, 256, 0, stream>>>(Sbuf, Pbuf, N);
    {
      dim3 g(D / 128, N / 128); // out = P @ V : (2048 x 512), V is (K,N) layout
      gemm_wmma<false, false, true, false><<<g, 256, 0, stream>>>(
          Pbuf, N, Vb, D, nullptr, Ob, D, N, D, N);
    }
  }
}
